// MoEFSCIL_24824910971120
// MI455X (gfx1250) — matrix operand order repaired and checked by reading, not compiled, not run on hardware
//
#include <hip/hip_runtime.h>
#include <hip/hip_bf16.h>

// ---------------------------------------------------------------------------
// MoE SS2D (VMamba-style) forward for MI455X / gfx1250.
// Heavy GEMMs use v_wmma_f32_16x16x32_bf16 with fragment-major packed
// operands: per (tile, k-step) each lane's 16 bf16 WMMA inputs are 32
// contiguous bytes -> LDS staging is b128 traffic and each fragment load is
// one 32B LDS read (2x ds_load_b128). Staging is double-buffered via the
// CDNA5 async global->LDS path (ASYNCcnt + s_wait_asynccnt) when available,
// else a register-pipelined load-early/store-late path.
// ---------------------------------------------------------------------------

#define BATCH 32
#define FS    7
#define LPOS  49            // FS*FS
#define DIM   512
#define NEXP  4
#define KD    4
#define NST   16
#define DTR   32
#define CDIM  64            // DT_RANK + 2*N_STATE
#define MROWS 1568          // BATCH*LPOS
#define N2D   1024          // 2*DIM

#define KT512 16            // K=512 -> 16 k-steps of 32
#define TILE_ELEMS 512      // 32 lanes * 16 bf16 per (16x32 or 32x16) tile

// GEMM1: A = x (1568x512) pad M->1600 (100 m-tiles); B = W (512x1024), 64 n-tiles
#define G1_MT 100
#define G1_NT 64
// GEMM2: A = xpw (64x512), 4 m-tiles; B = xs (512x1568) pad N->1664, 104 n-tiles
#define G2_MT 4
#define G2_NT 104
#define G2_NT_REAL 98       // 1568/16

#if __has_builtin(__builtin_amdgcn_global_load_async_to_lds_b128)
#define ASYNC_STAGE 1
#else
#define ASYNC_STAGE 0
#endif

typedef __attribute__((ext_vector_type(16))) __bf16 v16bf;
typedef __attribute__((ext_vector_type(8)))  float  v8f;
typedef int v4i_ __attribute__((vector_size(16)));                 // int4 vector
typedef __attribute__((address_space(1))) v4i_* gv4i_p;            // global int4*
typedef __attribute__((address_space(3))) v4i_* lv4i_p;            // LDS int4*

// ------------------------------ helpers ------------------------------------
__device__ __forceinline__ float silu_f(float x) {
    return x / (1.f + __expf(-x));
}
__device__ __forceinline__ float softplus_f(float x) {
    return (x > 20.f) ? x : log1pf(__expf(x));
}
// scan position t of direction k -> spatial index pos (row-major h*7+w).
// Maps are involutions, so the same function is also the inverse.
__device__ __forceinline__ int scan_pos(int k, int t) {
    int u = (k & 1) ? (LPOS - 1 - t) : t;
    if (k < 2) return u;
    return (u % FS) * FS + (u / FS);   // transpose h<->w
}
__device__ __forceinline__ float block_reduce_sum(float v, float* red) {
    int tid = threadIdx.x;
    red[tid] = v;
    __syncthreads();
    for (int s = blockDim.x >> 1; s > 0; s >>= 1) {
        if (tid < s) red[tid] += red[tid + s];
        __syncthreads();
    }
    float r = red[0];
    __syncthreads();
    return r;
}

// --------------------- operand packing (fragment-major) --------------------
// A-fragment mapping (16-bit A 16x32, contiguous k order, same as B): lane = half*16 + (m%16),
// element j -> k = 32*kt + 16*half + j.
__global__ void pack_a_kernel(const float* __restrict__ src, __bf16* __restrict__ dst,
                              int M, int K, int MT, int KT,
                              long sSrc, long sDst, int nb) {
    long total = (long)nb * MT * KT * TILE_ELEMS;
    long idx = (long)blockIdx.x * blockDim.x + threadIdx.x;
    long stride = (long)gridDim.x * blockDim.x;
    long per = (long)MT * KT * TILE_ELEMS;
    for (; idx < total; idx += stride) {
        long bz = idx / per;
        long r  = idx % per;
        int j    = (int)(r & 15);
        int lane = (int)((r >> 4) & 31);
        int t    = (int)(r >> 9);
        int kt   = t % KT;
        int mt   = t / KT;
        int row  = mt * 16 + (lane & 15);
        int half = lane >> 4;
        int k    = kt * 32 + 16 * half + j;
        float v  = (row < M) ? src[bz * sSrc + (size_t)row * K + k] : 0.f;
        dst[bz * sDst + r] = (__bf16)v;
    }
}

// B-fragment mapping (16-bit B 32x16): lane = half*16 + (n%16),
// element j -> k = 32*kt + 16*half + j.
__global__ void pack_b_kernel(const float* __restrict__ src, __bf16* __restrict__ dst,
                              int K, int N, int NT, int KT,
                              long sSrc, long sDst, int nb) {
    long total = (long)nb * NT * KT * TILE_ELEMS;
    long idx = (long)blockIdx.x * blockDim.x + threadIdx.x;
    long stride = (long)gridDim.x * blockDim.x;
    long per = (long)NT * KT * TILE_ELEMS;
    for (; idx < total; idx += stride) {
        long bz = idx / per;
        long r  = idx % per;
        int j    = (int)(r & 15);
        int lane = (int)((r >> 4) & 31);
        int t    = (int)(r >> 9);
        int kt   = t % KT;
        int nt   = t / KT;
        int col  = nt * 16 + (lane & 15);
        int half = lane >> 4;
        int k    = kt * 32 + 16 * half + j;
        float v  = (col < N) ? src[bz * sSrc + (size_t)k * N + col] : 0.f;
        dst[bz * sDst + r] = (__bf16)v;
    }
}

__global__ void fill_zero_bf16(__bf16* __restrict__ dst, long offsetPerB,
                               long strideB, long countPerB, int nb) {
    long total = (long)nb * countPerB;
    long idx = (long)blockIdx.x * blockDim.x + threadIdx.x;
    long stride = (long)gridDim.x * blockDim.x;
    for (; idx < total; idx += stride) {
        long bz = idx / countPerB, r = idx % countPerB;
        dst[bz * strideB + offsetPerB + r] = (__bf16)0.f;
    }
}

// ------------------------- bf16 WMMA GEMM (packed) -------------------------
// C(M x N) = A @ B [+ bias], batched over blockIdx.z. Block tile 64x128,
// 256 threads = 8 waves arranged 4(m) x 2(n); each wave: 16x64 strip,
// 4 v_wmma per k-step. Double-buffered LDS; async (or pipelined) staging.
__global__ __launch_bounds__(256) void gemm_bf16_wmma(
    const __bf16* __restrict__ Apk, long sAb,
    const __bf16* __restrict__ Bpk, long sBb,
    float* __restrict__ C, long sCb,
    const float* __restrict__ bias, long sBiasB,
    int M, int N, int KT) {
    __shared__ alignas(32) __bf16 lA[2][4 * TILE_ELEMS];   // 4 m-tiles
    __shared__ alignas(32) __bf16 lB[2][8 * TILE_ELEMS];   // 8 n-tiles

    const int bz = blockIdx.z;
    const __bf16* Ab = Apk + (size_t)bz * sAb;
    const __bf16* Bb = Bpk + (size_t)bz * sBb;
    float* Cb = C + (size_t)bz * sCb;

    const int tid  = threadIdx.x;
    const int lane = tid & 31;
    const int w    = tid >> 5;
    const int wm   = w >> 1;        // 0..3: wave m-subtile
    const int wn   = w & 1;         // 0..1: wave n-half
    const int mt0  = blockIdx.y * 4;
    const int nt0  = blockIdx.x * 8;

    // per-thread staging addresses: thread copies one b128 of A and two of B
    const int ch = tid >> 6;                 // chunk 0..3
    const int wi = (tid & 63) * 8;           // bf16 offset of this thread's b128
    const __bf16* gA  = Ab + ((size_t)(mt0 + ch) * KT) * TILE_ELEMS + wi;
    const __bf16* gB0 = Bb + ((size_t)(nt0 + ch) * KT) * TILE_ELEMS + wi;
    const __bf16* gB1 = Bb + ((size_t)(nt0 + 4 + ch) * KT) * TILE_ELEMS + wi;
    const int ldsA  = ch * TILE_ELEMS + wi;
    const int ldsB0 = ch * TILE_ELEMS + wi;
    const int ldsB1 = (4 + ch) * TILE_ELEMS + wi;

    v8f acc[4];
#pragma unroll
    for (int i = 0; i < 4; ++i)
        acc[i] = (v8f){0.f, 0.f, 0.f, 0.f, 0.f, 0.f, 0.f, 0.f};

#if ASYNC_STAGE
    // ---- async global->LDS staging (no VGPR round-trip, ASYNCcnt) ----
    auto stage_async = [&](int buf, int kt) {
        __builtin_amdgcn_global_load_async_to_lds_b128(
            (gv4i_p)(gA + (size_t)kt * TILE_ELEMS),
            (lv4i_p)(lA[buf] + ldsA), 0, 0);
        __builtin_amdgcn_global_load_async_to_lds_b128(
            (gv4i_p)(gB0 + (size_t)kt * TILE_ELEMS),
            (lv4i_p)(lB[buf] + ldsB0), 0, 0);
        __builtin_amdgcn_global_load_async_to_lds_b128(
            (gv4i_p)(gB1 + (size_t)kt * TILE_ELEMS),
            (lv4i_p)(lB[buf] + ldsB1), 0, 0);
    };
    stage_async(0, 0);
    for (int kt = 0; kt < KT; ++kt) {
        asm volatile("s_wait_asynccnt 0" ::: "memory");  // my LDS writes for kt landed
        __syncthreads();                                 // everyone's landed; prev buf free
        if (kt + 1 < KT) stage_async((kt + 1) & 1, kt + 1);  // DMA overlaps WMMAs
        const int buf = kt & 1;
        v16bf afrag = *(const v16bf*)(lA[buf] + (wm * 32 + lane) * 16);
#pragma unroll
        for (int nt = 0; nt < 4; ++nt) {
            v16bf bfrag = *(const v16bf*)(lB[buf] + ((wn * 4 + nt) * 32 + lane) * 16);
            acc[nt] = __builtin_amdgcn_wmma_f32_16x16x32_bf16(
                false, afrag, false, bfrag, (short)0, acc[nt], false, false);
        }
    }
#else
    // ---- register-pipelined staging: issue loads early, store after WMMAs --
    uint4 ra, rb0, rb1;
    auto stage_load = [&](int kt) {
        ra  = *(const uint4*)(gA  + (size_t)kt * TILE_ELEMS);
        rb0 = *(const uint4*)(gB0 + (size_t)kt * TILE_ELEMS);
        rb1 = *(const uint4*)(gB1 + (size_t)kt * TILE_ELEMS);
    };
    auto stage_store = [&](int buf) {
        *(uint4*)(lA[buf] + ldsA)  = ra;
        *(uint4*)(lB[buf] + ldsB0) = rb0;
        *(uint4*)(lB[buf] + ldsB1) = rb1;
    };
    stage_load(0);
    stage_store(0);
    for (int kt = 0; kt < KT; ++kt) {
        __syncthreads();                       // buf[kt] visible; prev buf free
        if (kt + 1 < KT) stage_load(kt + 1);   // global loads fly under the WMMAs
        const int buf = kt & 1;
        v16bf afrag = *(const v16bf*)(lA[buf] + (wm * 32 + lane) * 16);
#pragma unroll
        for (int nt = 0; nt < 4; ++nt) {
            v16bf bfrag = *(const v16bf*)(lB[buf] + ((wn * 4 + nt) * 32 + lane) * 16);
            acc[nt] = __builtin_amdgcn_wmma_f32_16x16x32_bf16(
                false, afrag, false, bfrag, (short)0, acc[nt], false, false);
        }
        if (kt + 1 < KT) stage_store((kt + 1) & 1);
    }
#endif

    // store: C/D 16x16 f32 layout: lane&15 = col, vgpr v -> row v + 8*(lane>>4)
    const int r = lane & 15, half = lane >> 4;
#pragma unroll
    for (int nt = 0; nt < 4; ++nt) {
#pragma unroll
        for (int v = 0; v < 8; ++v) {
            int mm = mt0 * 16 + wm * 16 + v + 8 * half;
            int nn = nt0 * 16 + (wn * 4 + nt) * 16 + r;
            if (mm < M && nn < N) {
                float out = acc[nt][v];
                if (bias) out += bias[(size_t)bz * sBiasB + nn];
                Cb[(size_t)mm * N + nn] = out;
            }
        }
    }
}

// ----------------------- gating: pool + softmax/top2 -----------------------
__global__ __launch_bounds__(256) void pool_x_kernel(const float* __restrict__ x,
                                                     float* __restrict__ x_flat) {
    int b = blockIdx.x;
    int d0 = threadIdx.x, d1 = threadIdx.x + 256;
    float s0 = 0.f, s1 = 0.f;
    const float* xb = x + (size_t)b * LPOS * DIM;
    for (int l = 0; l < LPOS; ++l) {
        s0 += xb[(size_t)l * DIM + d0];
        s1 += xb[(size_t)l * DIM + d1];
    }
    x_flat[(size_t)b * DIM + d0] = s0 * (1.f / LPOS);
    x_flat[(size_t)b * DIM + d1] = s1 * (1.f / LPOS);
}

__global__ __launch_bounds__(128) void gate_kernel(
    const float* __restrict__ x_flat, const float* __restrict__ w_gate,
    const float* __restrict__ b_gate, float* __restrict__ topv,
    int* __restrict__ topi, float* __restrict__ aux) {
    __shared__ float raw[BATCH][NEXP];
    __shared__ float msk[BATCH][NEXP];
    __shared__ float denom[NEXP];
    __shared__ float auxv[NEXP];
    int tid = threadIdx.x;
    int b = tid >> 2, e = tid & 3;
    float lg = b_gate[e];
    for (int dd = 0; dd < DIM; ++dd)
        lg += x_flat[(size_t)b * DIM + dd] * w_gate[dd * NEXP + e];
    raw[b][e] = lg;
    __syncthreads();
    if (e == 0) {
        float m = raw[b][0];
        for (int j = 1; j < NEXP; ++j) m = fmaxf(m, raw[b][j]);
        float ex[NEXP], s = 0.f;
        for (int j = 0; j < NEXP; ++j) { ex[j] = __expf(raw[b][j] - m); s += ex[j]; }
        for (int j = 0; j < NEXP; ++j) raw[b][j] = ex[j] / s;
        int i0 = 0;
        for (int j = 1; j < NEXP; ++j) if (raw[b][j] > raw[b][i0]) i0 = j;
        int i1 = (i0 == 0) ? 1 : 0;
        for (int j = 0; j < NEXP; ++j)
            if (j != i0 && raw[b][j] > raw[b][i1]) i1 = j;
        for (int j = 0; j < NEXP; ++j) msk[b][j] = (j == i0 || j == i1) ? 1.f : 0.f;
    }
    __syncthreads();
    if (tid < NEXP) {
        float dsum = 0.f, imp = 0.f, ld = 0.f;
        for (int bb = 0; bb < BATCH; ++bb) {
            dsum += raw[bb][tid] * msk[bb][tid];
            imp  += raw[bb][tid];
            ld   += msk[bb][tid];
        }
        denom[tid] = dsum + 1e-6f;
        imp *= (1.f / BATCH); ld *= (1.f / BATCH);
        auxv[tid] = (ld - imp) * (ld - imp);
    }
    __syncthreads();
    if (tid == 0)
        aux[0] = 0.01f * 0.25f * (auxv[0] + auxv[1] + auxv[2] + auxv[3]);
    if (tid < BATCH) {
        float gs[NEXP];
        for (int j = 0; j < NEXP; ++j)
            gs[j] = raw[tid][j] * msk[tid][j] / denom[j] * 40.f;  // capacity=int(1.25*32)
        int i0 = 0;
        for (int j = 1; j < NEXP; ++j) if (gs[j] > gs[i0]) i0 = j;
        int i1 = (i0 == 0) ? 1 : 0;
        for (int j = 0; j < NEXP; ++j) if (j != i0 && gs[j] > gs[i1]) i1 = j;
        topv[tid * 2 + 0] = gs[i0]; topi[tid * 2 + 0] = i0;
        topv[tid * 2 + 1] = gs[i1]; topi[tid * 2 + 1] = i1;
    }
}

// ------------- depthwise conv + SiLU + packed direction scatter ------------
__device__ __forceinline__ size_t xs_packed_idx(int n, int c) {
    int nt = n >> 4, nn = n & 15;
    int kt = c >> 5, half = (c >> 4) & 1, j = c & 15;
    int lane = half * 16 + nn;
    return ((size_t)(nt * KT512 + kt) * 32 + lane) * 16 + j;
}

__global__ __launch_bounds__(512) void conv_silu_scatter(
    const float* __restrict__ xz, const float* __restrict__ cw,
    const float* __restrict__ cb, float* __restrict__ xc,
    __bf16* __restrict__ xs) {
    int eb = blockIdx.x;
    int e = eb >> 5, b = eb & 31;
    int c = threadIdx.x;
    float wgt[9];
#pragma unroll
    for (int i = 0; i < 9; ++i) wgt[i] = cw[((size_t)(e * DIM + c)) * 9 + i];
    float bias = cb[e * DIM + c];
    const float* xzb = xz + ((size_t)e * MROWS + b * LPOS) * N2D;
    float* xcb = xc + ((size_t)(e * BATCH + b) * DIM + c) * LPOS;
    const size_t strideB = (size_t)G2_NT * KT512 * TILE_ELEMS;  // per (e,k) batch
    __bf16* xs_e = xs + (size_t)(e * KD) * strideB;
    for (int h = 0; h < FS; ++h) {
        for (int w = 0; w < FS; ++w) {
            float acc = bias;
#pragma unroll
            for (int dh = -1; dh <= 1; ++dh) {
                int hh = h + dh;
                if (hh < 0 || hh > 6) continue;
#pragma unroll
                for (int dw = -1; dw <= 1; ++dw) {
                    int ww = w + dw;
                    if (ww < 0 || ww > 6) continue;
                    acc += wgt[(dh + 1) * 3 + (dw + 1)] *
                           xzb[(size_t)(hh * FS + ww) * N2D + c];
                }
            }
            float v = silu_f(acc);
            int pos = h * FS + w;
            xcb[pos] = v;
            __bf16 vb = (__bf16)v;
            int t2 = (pos % FS) * FS + (pos / FS);
            int nb0 = b * LPOS;
            xs_e[xs_packed_idx(nb0 + pos, c)]                            = vb;
            xs_e[strideB     + xs_packed_idx(nb0 + (LPOS - 1 - pos), c)] = vb;
            xs_e[2 * strideB + xs_packed_idx(nb0 + t2, c)]               = vb;
            xs_e[3 * strideB + xs_packed_idx(nb0 + (LPOS - 1 - t2), c)]  = vb;
        }
    }
}

// ------------------------------ selective scan -----------------------------
__global__ __launch_bounds__(512) void scan_kernel(
    const float* __restrict__ x_dbl, const float* __restrict__ xc,
    const float* __restrict__ dtw, const float* __restrict__ dtb,
    const float* __restrict__ A_log, const float* __restrict__ Ds,
    float* __restrict__ ycomb) {
    __shared__ float xd[CDIM * LPOS];  // 12.25 KB
    int eb = blockIdx.x;
    int e = eb >> 5, b = eb & 31;
    int d = threadIdx.x;
    const float* xc_bd = xc + ((size_t)(e * BATCH + b) * DIM + d) * LPOS;
    float* y_bd = ycomb + ((size_t)(e * BATCH + b) * DIM + d) * LPOS;

    for (int k = 0; k < KD; ++k) {
        const float* xdb = x_dbl + ((size_t)(e * KD + k) * CDIM) * MROWS + (size_t)b * LPOS;
        for (int i = d; i < CDIM * LPOS; i += 512) {
            int c = i / LPOS, t = i % LPOS;
            xd[i] = xdb[(size_t)c * MROWS + t];
        }
        __syncthreads();

        const size_t ekd = (size_t)(e * KD + k) * DIM + d;
        float dtw_r[DTR];
#pragma unroll
        for (int r = 0; r < DTR; ++r) dtw_r[r] = dtw[ekd * DTR + r];
        float Arow[NST];
#pragma unroll
        for (int n = 0; n < NST; ++n) Arow[n] = -__expf(A_log[ekd * NST + n]);
        float dtbv = dtb[ekd];
        float Dsv  = Ds[ekd];

        float h[NST];
#pragma unroll
        for (int n = 0; n < NST; ++n) h[n] = 0.f;

        for (int t = 0; t < LPOS; ++t) {
            int pos = scan_pos(k, t);
            float x_t = xc_bd[pos];
            float s = dtbv;
#pragma unroll
            for (int r = 0; r < DTR; ++r) s += dtw_r[r] * xd[r * LPOS + t];
            float dt = softplus_f(s);
            float dx = dt * x_t;
            float y = 0.f;
#pragma unroll
            for (int n = 0; n < NST; ++n) {
                float Bn = xd[(DTR + n) * LPOS + t];
                float Cn = xd[(DTR + NST + n) * LPOS + t];
                h[n] = h[n] * __expf(dt * Arow[n]) + dx * Bn;
                y += h[n] * Cn;
            }
            float contrib = y + Dsv * x_t;
            if (k == 0) y_bd[pos] = contrib;
            else        y_bd[pos] += contrib;
        }
        __syncthreads();
    }
}

// ------------------ per-position LN * silu(z), pool, final LN --------------
__global__ __launch_bounds__(256) void finalize_expert(
    const float* __restrict__ ycomb, const float* __restrict__ xz,
    const float* __restrict__ og, const float* __restrict__ ob,
    const float* __restrict__ ng, const float* __restrict__ nb,
    float* __restrict__ all_out) {
    __shared__ float red[256];
    int eb = blockIdx.x;
    int e = eb >> 5, b = eb & 31;
    int tid = threadIdx.x;
    int d0 = tid, d1 = tid + 256;
    const float* yb = ycomb + ((size_t)(e * BATCH + b) * DIM) * LPOS;
    const float* zb = xz + ((size_t)e * MROWS + b * LPOS) * N2D + DIM;
    float acc0 = 0.f, acc1 = 0.f;
    for (int l = 0; l < LPOS; ++l) {
        float v0 = yb[(size_t)d0 * LPOS + l];
        float v1 = yb[(size_t)d1 * LPOS + l];
        float s  = block_reduce_sum(v0 + v1, red);
        float ss = block_reduce_sum(v0 * v0 + v1 * v1, red);
        float mu = s * (1.f / DIM);
        float var = ss * (1.f / DIM) - mu * mu;
        float rstd = rsqrtf(var + 1e-5f);
        float yn0 = (v0 - mu) * rstd * og[e * DIM + d0] + ob[e * DIM + d0];
        float yn1 = (v1 - mu) * rstd * og[e * DIM + d1] + ob[e * DIM + d1];
        float z0 = zb[(size_t)l * N2D + d0];
        float z1 = zb[(size_t)l * N2D + d1];
        acc0 += yn0 * silu_f(z0);
        acc1 += yn1 * silu_f(z1);
    }
    float p0 = acc0 * (1.f / LPOS), p1 = acc1 * (1.f / LPOS);
    float s  = block_reduce_sum(p0 + p1, red);
    float ss = block_reduce_sum(p0 * p0 + p1 * p1, red);
    float mu = s * (1.f / DIM);
    float var = ss * (1.f / DIM) - mu * mu;
    float rstd = rsqrtf(var + 1e-5f);
    float* ao = all_out + ((size_t)(e * BATCH + b)) * DIM;
    ao[d0] = (p0 - mu) * rstd * ng[e * DIM + d0] + nb[e * DIM + d0];
    ao[d1] = (p1 - mu) * rstd * ng[e * DIM + d1] + nb[e * DIM + d1];
}

// ------------------------------- mixing ------------------------------------
__global__ __launch_bounds__(256) void mix_kernel(
    const float* __restrict__ all_out, const float* __restrict__ topv,
    const int* __restrict__ topi, const float* __restrict__ aux,
    float* __restrict__ out) {
    int idx = blockIdx.x * blockDim.x + threadIdx.x;
    if (idx < BATCH * DIM) {
        int b = idx >> 9, d = idx & 511;
        float s = 0.f;
#pragma unroll
        for (int j = 0; j < 2; ++j) {
            int ei = topi[b * 2 + j];
            s += topv[b * 2 + j] * all_out[((size_t)ei * BATCH + b) * DIM + d];
        }
        out[idx] = s;
    }
    if (idx == 0) out[BATCH * DIM] = aux[0];
}

// ------------------------------- launcher ----------------------------------
extern "C" void kernel_launch(void* const* d_in, const int* in_sizes, int n_in,
                              void* d_out, int out_size, void* d_ws, size_t ws_size,
                              hipStream_t stream) {
    const float* x        = (const float*)d_in[0];
    const float* w_gate   = (const float*)d_in[1];
    const float* b_gate   = (const float*)d_in[2];
    const float* in_w     = (const float*)d_in[3];
    const float* in_b     = (const float*)d_in[4];
    const float* conv_w   = (const float*)d_in[5];
    const float* conv_b   = (const float*)d_in[6];
    const float* xpw      = (const float*)d_in[7];
    const float* dtw      = (const float*)d_in[8];
    const float* dtb      = (const float*)d_in[9];
    const float* A_log    = (const float*)d_in[10];
    const float* Ds       = (const float*)d_in[11];
    const float* og       = (const float*)d_in[12];
    const float* ob       = (const float*)d_in[13];
    const float* ng       = (const float*)d_in[14];
    const float* nb       = (const float*)d_in[15];
    float* out = (float*)d_out;

    char* ws = (char*)d_ws;
    size_t off = 0;
    auto take = [&](size_t bytes) {
        char* p = ws + off;
        off = (off + bytes + 255) & ~(size_t)255;
        return p;
    };
    const size_t xpk_sz  = (size_t)G1_MT * KT512 * TILE_ELEMS;           // x packed-A
    const size_t wpk_sz  = (size_t)G1_NT * KT512 * TILE_ELEMS;           // per-e W packed-B
    const size_t xpwpk_sz= (size_t)G2_MT * KT512 * TILE_ELEMS;           // per-(e,k) xpw packed-A
    const size_t xspk_sz = (size_t)G2_NT * KT512 * TILE_ELEMS;           // per-(e,k) xs packed-B

    float*  x_flat  = (float*) take((size_t)BATCH * DIM * 4);
    float*  topv    = (float*) take((size_t)BATCH * 2 * 4);
    int*    topi    = (int*)   take((size_t)BATCH * 2 * 4);
    float*  auxp    = (float*) take(4);
    __bf16* x_pk    = (__bf16*)take(xpk_sz * 2);
    __bf16* w_pk    = (__bf16*)take((size_t)NEXP * wpk_sz * 2);
    __bf16* xpw_pk  = (__bf16*)take((size_t)NEXP * KD * xpwpk_sz * 2);
    float*  xz      = (float*) take((size_t)NEXP * MROWS * N2D * 4);
    float*  xc      = (float*) take((size_t)NEXP * BATCH * DIM * LPOS * 4);
    __bf16* xs_pk   = (__bf16*)take((size_t)NEXP * KD * xspk_sz * 2);
    float*  x_dbl   = (float*) take((size_t)NEXP * KD * CDIM * MROWS * 4);
    float*  ycomb   = (float*) take((size_t)NEXP * BATCH * DIM * LPOS * 4);
    float*  all_out = (float*) take((size_t)NEXP * BATCH * DIM * 4);
    (void)ws_size; (void)in_sizes; (void)n_in; (void)out_size;

    // 1) pack GEMM operands (f32 -> bf16, fragment-major)
    pack_a_kernel<<<2048, 256, 0, stream>>>(x, x_pk, MROWS, DIM, G1_MT, KT512,
                                            0L, (long)xpk_sz, 1);
    pack_b_kernel<<<4096, 256, 0, stream>>>(in_w, w_pk, DIM, N2D, G1_NT, KT512,
                                            (long)DIM * N2D, (long)wpk_sz, NEXP);
    pack_a_kernel<<<1024, 256, 0, stream>>>(xpw, xpw_pk, CDIM, DIM, G2_MT, KT512,
                                            (long)CDIM * DIM, (long)xpwpk_sz, NEXP * KD);
    // zero the padded n-tile tail of xs (tiles 98..103 per (e,k))
    fill_zero_bf16<<<1536, 256, 0, stream>>>(
        xs_pk, (long)G2_NT_REAL * KT512 * TILE_ELEMS, (long)xspk_sz,
        (long)(G2_NT - G2_NT_REAL) * KT512 * TILE_ELEMS, NEXP * KD);

    // 2) gating
    pool_x_kernel<<<BATCH, 256, 0, stream>>>(x, x_flat);
    gate_kernel<<<1, 128, 0, stream>>>(x_flat, w_gate, b_gate, topv, topi, auxp);

    // 3) in_proj GEMM: xz[e] = x(1568x512) @ W[e](512x1024) + b[e]
    {
        dim3 grid(G1_NT / 8, G1_MT / 4, NEXP);   // 8 x 25 x 4
        gemm_bf16_wmma<<<grid, 256, 0, stream>>>(
            x_pk, 0L, w_pk, (long)wpk_sz,
            xz, (long)MROWS * N2D, in_b, (long)N2D, MROWS, N2D, KT512);
    }

    // 4) depthwise conv + SiLU + packed direction scatter
    conv_silu_scatter<<<NEXP * BATCH, 512, 0, stream>>>(xz, conv_w, conv_b, xc, xs_pk);

    // 5) x_dbl GEMM per (e,k): (64x512) @ (512x1568)
    {
        dim3 grid(G2_NT / 8, G2_MT / 4, NEXP * KD);  // 13 x 1 x 16
        gemm_bf16_wmma<<<grid, 256, 0, stream>>>(
            xpw_pk, (long)xpwpk_sz, xs_pk, (long)xspk_sz,
            x_dbl, (long)CDIM * MROWS, nullptr, 0L, CDIM, MROWS, KT512);
    }

    // 6) selective scan (4 directions combined into ycomb)
    scan_kernel<<<NEXP * BATCH, 512, 0, stream>>>(x_dbl, xc, dtw, dtb, A_log, Ds, ycomb);

    // 7) LN * silu(z), pool over L, final LN -> all_out
    finalize_expert<<<NEXP * BATCH, 256, 0, stream>>>(ycomb, xz, og, ob, ng, nb, all_out);

    // 8) top-2 mixing + aux loss
    mix_kernel<<<(BATCH * DIM + 255) / 256, 256, 0, stream>>>(all_out, topv, topi, auxp, out);
}
